// EncoderLayer_89558658056554
// MI455X (gfx1250) — matrix operand order repaired and checked by reading, not compiled, not run on hardware
//
#include <hip/hip_runtime.h>

// ---------------------------------------------------------------------------
// Types for CDNA5 WMMA (wave32): v_wmma_f32_16x16x32_bf16
// ---------------------------------------------------------------------------
typedef __attribute__((ext_vector_type(16))) __bf16 v16bf;
typedef __attribute__((ext_vector_type(8)))  __bf16 v8bf;
typedef __attribute__((ext_vector_type(8)))  float  v8f;
typedef __attribute__((ext_vector_type(4)))  int    v4i;

static __device__ __forceinline__ v16bf cat8(v8bf lo, v8bf hi) {
  return __builtin_shufflevector(lo, hi, 0,1,2,3,4,5,6,7,8,9,10,11,12,13,14,15);
}

static __device__ __forceinline__ v8f wmma_bf16(v16bf a, v16bf b, v8f c) {
  // (neg_a, A, neg_b, B, c_mod, C, reuse_a, reuse_b)
  return __builtin_amdgcn_wmma_f32_16x16x32_bf16(false, a, false, b, (short)0, c,
                                                 false, false);
}

// CDNA5 async global->LDS copies (ASYNCcnt tracked), guarded for portability.
#if __has_builtin(__builtin_amdgcn_global_load_async_to_lds_b128) && \
    __has_builtin(__builtin_amdgcn_s_wait_asynccnt)
#define USE_ASYNC_COPY 1
typedef __attribute__((address_space(1))) v4i* gptr_b128;
typedef __attribute__((address_space(3))) v4i* lptr_b128;
static __device__ __forceinline__ void async_copy16(void* lds, const void* g) {
  __builtin_amdgcn_global_load_async_to_lds_b128((gptr_b128)g, (lptr_b128)lds,
                                                 0, 0);
}
static __device__ __forceinline__ void async_wait_all() {
  __builtin_amdgcn_s_wait_asynccnt(0);
}
#else
#define USE_ASYNC_COPY 0
static __device__ __forceinline__ void async_wait_all() {}
#endif

// Problem constants
#define BATCH 4
#define SEQ   2048
#define DMODEL 1024
#define NHEAD 16
#define DHEAD 64
#define DFF   4096
#define BSROWS (BATCH * SEQ)   // 8192

// ---------------------------------------------------------------------------
// f32 -> bf16 conversion (activations, row-major preserved)
// ---------------------------------------------------------------------------
__global__ void f32_to_bf16_kernel(const float* __restrict__ in,
                                   __bf16* __restrict__ out, int n) {
  int i = blockIdx.x * blockDim.x + threadIdx.x;
  int stride = gridDim.x * blockDim.x;
  for (; i < n; i += stride) out[i] = (__bf16)in[i];
}

// ---------------------------------------------------------------------------
// f32 [K][N] -> bf16 [N][K] transposed conversion (weights, one-time).
// 32x32 tiles via LDS, block = 256 threads (32x8).
// ---------------------------------------------------------------------------
__global__ __launch_bounds__(256) void transpose_to_bf16_kernel(
    const float* __restrict__ in, __bf16* __restrict__ out, int K, int N) {
  __shared__ float tile[32][33];
  const int k0 = blockIdx.y * 32, n0 = blockIdx.x * 32;
  const int tx = threadIdx.x & 31, ty = threadIdx.x >> 5;
#pragma unroll
  for (int i = 0; i < 32; i += 8)
    tile[ty + i][tx] = in[(size_t)(k0 + ty + i) * N + n0 + tx];
  __syncthreads();
#pragma unroll
  for (int i = 0; i < 32; i += 8)
    out[(size_t)(n0 + ty + i) * K + k0 + tx] = (__bf16)tile[tx][ty + i];
}

// ---------------------------------------------------------------------------
// bf16 GEMM: C[M,N] = A[M,K] @ B[K,N] + bias.  B is provided TRANSPOSED
// (Bt[N][K]) so both LDS tiles are contiguous-K b128 (async-able) copies.
// Block tile 128x128, BK=32, 8 waves (4 M x 2 N), wave tile 32x64.
// Double-buffered LDS staging.
// MODE: 0 = f32 row-major, 1 = bf16 row-major, 2 = bf16 transposed [N][M].
// ---------------------------------------------------------------------------
#define LDS_STRIDE 40   // padded bf16 stride; 80 bytes -> 16B-aligned rows

template <int MODE, int RELU>
__global__ __launch_bounds__(256) void gemm_bf16_kernel(
    const __bf16* __restrict__ A, const __bf16* __restrict__ Bt,
    const float* __restrict__ bias, void* __restrict__ Cptr,
    int M, int N, int K) {
  __shared__ alignas(16) __bf16 As[2][128 * LDS_STRIDE];   // As[m][k]
  __shared__ alignas(16) __bf16 Bs[2][128 * LDS_STRIDE];   // Bs[n][k]

  const int tid  = threadIdx.x;
  const int lane = tid & 31;
  const int wave = tid >> 5;
  const int l16  = lane & 15;
  const int half = lane >> 4;
  const int wm   = (wave & 3) * 32;   // wave M offset inside block
  const int wn   = (wave >> 2) * 64;  // wave N offset inside block
  const int m0   = blockIdx.y * 128;
  const int n0   = blockIdx.x * 128;

  // Staging addresses: thread t copies 16 contiguous bf16 for A and for Bt.
  const int sr = tid >> 1, sc = (tid & 1) * 16;
  const __bf16* Asrc = A  + (size_t)(m0 + sr) * K + sc;
  const __bf16* Bsrc = Bt + (size_t)(n0 + sr) * K + sc;
  const int soff = sr * LDS_STRIDE + sc;

  v8f acc[2][4] = {};
  const int nk = K >> 5;

  auto stage = [&](int buf, int k0) {
#if USE_ASYNC_COPY
    async_copy16(&As[buf][soff],     Asrc + k0);
    async_copy16(&As[buf][soff + 8], Asrc + k0 + 8);
    async_copy16(&Bs[buf][soff],     Bsrc + k0);
    async_copy16(&Bs[buf][soff + 8], Bsrc + k0 + 8);
#else
    v8bf a0 = *(const v8bf*)(Asrc + k0);
    v8bf a1 = *(const v8bf*)(Asrc + k0 + 8);
    v8bf b0 = *(const v8bf*)(Bsrc + k0);
    v8bf b1 = *(const v8bf*)(Bsrc + k0 + 8);
    *(v8bf*)&As[buf][soff]     = a0;
    *(v8bf*)&As[buf][soff + 8] = a1;
    *(v8bf*)&Bs[buf][soff]     = b0;
    *(v8bf*)&Bs[buf][soff + 8] = b1;
#endif
  };

  stage(0, 0);
  for (int kt = 0; kt < nk; ++kt) {
    const int buf = kt & 1;
    async_wait_all();
    __syncthreads();
    if (kt + 1 < nk) stage(buf ^ 1, (kt + 1) << 5);

    v16bf afrag[2], bfrag[4];
#pragma unroll
    for (int i = 0; i < 2; ++i) {
      const __bf16* p = &As[buf][(wm + i * 16 + l16) * LDS_STRIDE];
      v8bf lo = *(const v8bf*)(p + 8 * half);        // K = 8h .. 8h+7
      v8bf hi = *(const v8bf*)(p + 16 + 8 * half);   // K = 16+8h .. +7
      afrag[i] = cat8(lo, hi);
    }
#pragma unroll
    for (int j = 0; j < 4; ++j) {
      const __bf16* p = &Bs[buf][(wn + j * 16 + l16) * LDS_STRIDE];
      v8bf lo = *(const v8bf*)(p + 8 * half);        // K = 8h .. 8h+7
      v8bf hi = *(const v8bf*)(p + 16 + 8 * half);   // K = 16+8h .. +7
      bfrag[j] = cat8(lo, hi);
    }
#pragma unroll
    for (int i = 0; i < 2; ++i)
#pragma unroll
      for (int j = 0; j < 4; ++j)
        acc[i][j] = wmma_bf16(afrag[i], bfrag[j], acc[i][j]);
  }

  // --- epilogue: bias (+relu), compile-time output mode ---
  float* Cf = (float*)Cptr;
  __bf16* Cb = (__bf16*)Cptr;
#pragma unroll
  for (int i = 0; i < 2; ++i) {
#pragma unroll
    for (int j = 0; j < 4; ++j) {
      const int col = n0 + wn + j * 16 + l16;
      const float bv = bias[col];
#pragma unroll
      for (int r = 0; r < 8; ++r) {
        const int row = m0 + wm + i * 16 + half * 8 + r;  // C: M = vgpr + 8h
        float v = acc[i][j][r] + bv;
        if (RELU) v = fmaxf(v, 0.0f);
        if (MODE == 0)      Cf[(size_t)row * N + col] = v;
        else if (MODE == 1) Cb[(size_t)row * N + col] = (__bf16)v;
        else                Cb[(size_t)col * M + row] = (__bf16)v;  // [N][M]
      }
    }
  }
}

// ---------------------------------------------------------------------------
// Flash attention. Q/K bf16 [B*S, D] (head h = cols h*64..h*64+63),
// V bf16 TRANSPOSED [D, B*S].  grid(S/128, H, B), block 256 (8 waves),
// wave owns 16 query rows.  Only LDS use: wave-private P tile (no barriers).
// ---------------------------------------------------------------------------
__global__ __launch_bounds__(256) void attn_kernel(
    const __bf16* __restrict__ Qb, const __bf16* __restrict__ Kb,
    const __bf16* __restrict__ Vt, __bf16* __restrict__ Ob) {
  __shared__ alignas(16) __bf16 Pw[8][16 * LDS_STRIDE];  // per-wave P tile

  const int tid  = threadIdx.x;
  const int lane = tid & 31;
  const int wave = tid >> 5;
  const int l16  = lane & 15;
  const int half = lane >> 4;
  const int h = blockIdx.y;
  const int b = blockIdx.z;
  const size_t rowbase = (size_t)b * SEQ;
  const int qrow = blockIdx.x * 128 + wave * 16;
  const int hcol = h * DHEAD;

  // Q A-fragments (16 rows x 64 cols -> 2 frags of 16x32), direct from global.
  v16bf qf[2];
  {
    const __bf16* qp = Qb + (rowbase + qrow + l16) * DMODEL + hcol;
#pragma unroll
    for (int i = 0; i < 2; ++i) {
      v8bf lo = *(const v8bf*)(qp + i * 32 + 8 * half);
      v8bf hi = *(const v8bf*)(qp + i * 32 + 16 + 8 * half);
      qf[i] = cat8(lo, hi);
    }
  }

  v8f o[4] = {};
  float mrun[8], lrun[8];
#pragma unroll
  for (int r = 0; r < 8; ++r) { mrun[r] = -3.0e38f; lrun[r] = 0.0f; }

  for (int kb = 0; kb < SEQ; kb += 32) {
    // Scores: 16x32 = Q(16x64) @ K_block^T(64x32); K^T frags straight from
    // global since K is row-major: b[e] = K[key n][kc + table-order k(e, half)].
    v8f sc[2] = {};
#pragma unroll
    for (int kt = 0; kt < 2; ++kt) {
      const __bf16* kp = Kb + (rowbase + kb + kt * 16 + l16) * DMODEL + hcol;
#pragma unroll
      for (int i = 0; i < 2; ++i) {
        v8bf lo = *(const v8bf*)(kp + i * 32 + 8 * half);
        v8bf hi = *(const v8bf*)(kp + i * 32 + 16 + 8 * half);
        sc[kt] = wmma_bf16(qf[i], cat8(lo, hi), sc[kt]);
      }
    }

    // Online softmax update (per row r; reduce across the 16 lanes of half)
#pragma unroll
    for (int r = 0; r < 8; ++r) {
      float s0 = sc[0][r] * 0.125f;   // 1/sqrt(64)
      float s1 = sc[1][r] * 0.125f;
      float mx = fmaxf(s0, s1);
#pragma unroll
      for (int off = 1; off < 16; off <<= 1)
        mx = fmaxf(mx, __shfl_xor(mx, off, 32));
      float nm = fmaxf(mrun[r], mx);
      float a  = __expf(mrun[r] - nm);
      float p0 = __expf(s0 - nm);
      float p1 = __expf(s1 - nm);
      float ps = p0 + p1;
#pragma unroll
      for (int off = 1; off < 16; off <<= 1) ps += __shfl_xor(ps, off, 32);
      lrun[r] = lrun[r] * a + ps;
      mrun[r] = nm;
      sc[0][r] = p0;
      sc[1][r] = p1;
#pragma unroll
      for (int t = 0; t < 4; ++t) o[t][r] *= a;
    }

    // P round-trip through wave-private LDS: C-layout -> A-fragment layout.
    {
      __bf16* prow = &Pw[wave][0];
#pragma unroll
      for (int r = 0; r < 8; ++r) {
        int m = r + 8 * half;
        prow[m * LDS_STRIDE + l16]      = (__bf16)sc[0][r];
        prow[m * LDS_STRIDE + 16 + l16] = (__bf16)sc[1][r];
      }
    }
    v16bf pf;
    {
      const __bf16* p = &Pw[wave][l16 * LDS_STRIDE];
      v8bf lo = *(const v8bf*)(p + 8 * half);
      v8bf hi = *(const v8bf*)(p + 16 + 8 * half);
      pf = cat8(lo, hi);
    }

    // V B-fragments straight from global transposed V: b[e] = Vt[d][key].
#pragma unroll
    for (int t = 0; t < 4; ++t) {
      const __bf16* vp =
          Vt + (size_t)(hcol + t * 16 + l16) * BSROWS + rowbase + kb;
      v8bf lo = *(const v8bf*)(vp + 8 * half);
      v8bf hi = *(const v8bf*)(vp + 16 + 8 * half);
      o[t] = wmma_bf16(pf, cat8(lo, hi), o[t]);
    }
  }

  // Epilogue: O / l, write bf16 into attn buffer [B*S, D]
#pragma unroll
  for (int t = 0; t < 4; ++t) {
#pragma unroll
    for (int r = 0; r < 8; ++r) {
      int m = r + 8 * half;
      size_t idx = (rowbase + qrow + m) * (size_t)DMODEL + hcol + t * 16 + l16;
      Ob[idx] = (__bf16)(o[t][r] / lrun[r]);
    }
  }
}

// ---------------------------------------------------------------------------
// LayerNorm over rows of 1024. Optional residual R; writes fp32 or bf16.
// Safe for in-place (X == outp): all reads of the row happen before stores.
// ---------------------------------------------------------------------------
template <int HAS_RES, int OUT_BF16>
__global__ __launch_bounds__(256) void ln_kernel(
    const float* __restrict__ X, const float* __restrict__ R,
    const float* __restrict__ gamma, const float* __restrict__ beta,
    void* __restrict__ outp) {
  const int D = DMODEL;
  const int row = blockIdx.x;
  const int tid = threadIdx.x;
  __shared__ float red[256];

  const float4* X4 = (const float4*)(X + (size_t)row * D);
  float4 v = X4[tid];
  if (HAS_RES) {
    const float4* R4 = (const float4*)(R + (size_t)row * D);
    float4 rv = R4[tid];
    v.x += rv.x; v.y += rv.y; v.z += rv.z; v.w += rv.w;
  }
  red[tid] = v.x + v.y + v.z + v.w;
  __syncthreads();
  for (int st = 128; st > 0; st >>= 1) {
    if (tid < st) red[tid] += red[tid + st];
    __syncthreads();
  }
  float mean = red[0] * (1.0f / D);
  __syncthreads();

  float dx = v.x - mean, dy = v.y - mean, dz = v.z - mean, dw = v.w - mean;
  red[tid] = dx * dx + dy * dy + dz * dz + dw * dw;
  __syncthreads();
  for (int st = 128; st > 0; st >>= 1) {
    if (tid < st) red[tid] += red[tid + st];
    __syncthreads();
  }
  float rs = rsqrtf(red[0] * (1.0f / D) + 1e-5f);

  float4 g = ((const float4*)gamma)[tid];
  float4 bb = ((const float4*)beta)[tid];
  float4 y;
  y.x = dx * rs * g.x + bb.x;
  y.y = dy * rs * g.y + bb.y;
  y.z = dz * rs * g.z + bb.z;
  y.w = dw * rs * g.w + bb.w;
  if (OUT_BF16) {
    __bf16* ob = (__bf16*)outp + (size_t)row * D + tid * 4;
    ob[0] = (__bf16)y.x; ob[1] = (__bf16)y.y;
    ob[2] = (__bf16)y.z; ob[3] = (__bf16)y.w;
  } else {
    ((float4*)((float*)outp + (size_t)row * D))[tid] = y;
  }
}

// ---------------------------------------------------------------------------
// Host launcher
// ---------------------------------------------------------------------------
extern "C" void kernel_launch(void* const* d_in, const int* in_sizes, int n_in,
                              void* d_out, int out_size, void* d_ws,
                              size_t ws_size, hipStream_t stream) {
  const float* x     = (const float*)d_in[0];
  const float* Wq    = (const float*)d_in[1];
  const float* bq    = (const float*)d_in[2];
  const float* Wk    = (const float*)d_in[3];
  const float* bk    = (const float*)d_in[4];
  const float* Wv    = (const float*)d_in[5];
  const float* bv    = (const float*)d_in[6];
  const float* Wfc   = (const float*)d_in[7];
  const float* bfc   = (const float*)d_in[8];
  const float* ln1_g = (const float*)d_in[9];
  const float* ln1_b = (const float*)d_in[10];
  const float* W1    = (const float*)d_in[11];
  const float* b1    = (const float*)d_in[12];
  const float* W2    = (const float*)d_in[13];
  const float* b2    = (const float*)d_in[14];
  const float* ln2_g = (const float*)d_in[15];
  const float* ln2_b = (const float*)d_in[16];
  float* out = (float*)d_out;

  // ---- workspace carve ----
  char* w = (char*)d_ws;
  size_t off = 0;
  auto carve = [&](size_t bytes) -> void* {
    void* p = w + off;
    off += (bytes + 255) & ~(size_t)255;
    return p;
  };
  __bf16* xb    = (__bf16*)carve((size_t)BSROWS * DMODEL * 2);
  __bf16* wqb   = (__bf16*)carve((size_t)DMODEL * DMODEL * 2);  // transposed
  __bf16* wkb   = (__bf16*)carve((size_t)DMODEL * DMODEL * 2);  // transposed
  __bf16* wvb   = (__bf16*)carve((size_t)DMODEL * DMODEL * 2);  // transposed
  __bf16* wfcb  = (__bf16*)carve((size_t)DMODEL * DMODEL * 2);  // transposed
  __bf16* w1b   = (__bf16*)carve((size_t)DMODEL * DFF * 2);     // transposed
  __bf16* w2b   = (__bf16*)carve((size_t)DFF * DMODEL * 2);     // transposed
  __bf16* qbuf  = (__bf16*)carve((size_t)BSROWS * DMODEL * 2);  // [BS][D]
  __bf16* kbuf  = (__bf16*)carve((size_t)BSROWS * DMODEL * 2);  // [BS][D]
  __bf16* vtbuf = (__bf16*)carve((size_t)BSROWS * DMODEL * 2);  // [D][BS]
  __bf16* attnb = (__bf16*)carve((size_t)BSROWS * DMODEL * 2);  // [BS][D]
  __bf16* hb    = (__bf16*)carve((size_t)BSROWS * DMODEL * 2);  // [BS][D]
  __bf16* m1b   = (__bf16*)carve((size_t)BSROWS * DFF * 2);     // [BS][DFF]
  (void)ws_size; (void)in_sizes; (void)n_in; (void)out_size;

  // ---- 1. conversions: x plain, weights transposed to [N][K] bf16 ----
  f32_to_bf16_kernel<<<2048, 256, 0, stream>>>(x, xb, BSROWS * DMODEL);
  dim3 blk(256);
  dim3 gTd(DMODEL / 32, DMODEL / 32);
  transpose_to_bf16_kernel<<<gTd, blk, 0, stream>>>(Wq,  wqb,  DMODEL, DMODEL);
  transpose_to_bf16_kernel<<<gTd, blk, 0, stream>>>(Wk,  wkb,  DMODEL, DMODEL);
  transpose_to_bf16_kernel<<<gTd, blk, 0, stream>>>(Wv,  wvb,  DMODEL, DMODEL);
  transpose_to_bf16_kernel<<<gTd, blk, 0, stream>>>(Wfc, wfcb, DMODEL, DMODEL);
  dim3 gT1(DFF / 32, DMODEL / 32);
  transpose_to_bf16_kernel<<<gT1, blk, 0, stream>>>(W1, w1b, DMODEL, DFF);
  dim3 gT2(DMODEL / 32, DFF / 32);
  transpose_to_bf16_kernel<<<gT2, blk, 0, stream>>>(W2, w2b, DFF, DMODEL);

  // ---- 2. QKV projections ----
  dim3 gD(DMODEL / 128, BSROWS / 128);
  gemm_bf16_kernel<1, 0><<<gD, blk, 0, stream>>>(xb, wqb, bq, qbuf,
                                                 BSROWS, DMODEL, DMODEL);
  gemm_bf16_kernel<1, 0><<<gD, blk, 0, stream>>>(xb, wkb, bk, kbuf,
                                                 BSROWS, DMODEL, DMODEL);
  gemm_bf16_kernel<2, 0><<<gD, blk, 0, stream>>>(xb, wvb, bv, vtbuf,
                                                 BSROWS, DMODEL, DMODEL);

  // ---- 3. flash attention ----
  dim3 gA(SEQ / 128, NHEAD, BATCH);
  attn_kernel<<<gA, blk, 0, stream>>>(qbuf, kbuf, vtbuf, attnb);

  // ---- 4. output projection (fp32 into d_out as scratch) ----
  gemm_bf16_kernel<0, 0><<<gD, blk, 0, stream>>>(attnb, wfcb, bfc, out,
                                                 BSROWS, DMODEL, DMODEL);

  // ---- 5. h = LN(x + proj) -> bf16 ----
  ln_kernel<1, 1><<<BSROWS, 256, 0, stream>>>(out, x, ln1_g, ln1_b, hb);

  // ---- 6. MLP up + relu (bf16 out) ----
  dim3 gF(DFF / 128, BSROWS / 128);
  gemm_bf16_kernel<1, 1><<<gF, blk, 0, stream>>>(hb, w1b, b1, m1b,
                                                 BSROWS, DFF, DMODEL);

  // ---- 7. MLP down (fp32 into d_out) ----
  gemm_bf16_kernel<0, 0><<<gD, blk, 0, stream>>>(m1b, w2b, b2, out,
                                                 BSROWS, DMODEL, DFF);

  // ---- 8. out = LN(m) in-place ----
  ln_kernel<0, 0><<<BSROWS, 256, 0, stream>>>(out, nullptr, ln2_g, ln2_b, out);
}
